// GaussianRenderer_8615704395801
// MI455X (gfx1250) — matrix operand order repaired and checked by reading, not compiled, not run on hardware
//
#include <hip/hip_runtime.h>
#include <hip/hip_bf16.h>

typedef __attribute__((ext_vector_type(16))) _Float16 v16h;
typedef __attribute__((ext_vector_type(8)))  float    v8f;

#define BEV_W 200
#define NPIX  40000
#define NG    2000
#define ND    32
#define NB    2

__device__ __forceinline__ unsigned pack2h(float a, float b) {
  union { _Float16 h; unsigned short s; } ua, ub;
  ua.h = (_Float16)a;
  ub.h = (_Float16)b;
  return (unsigned)ua.s | ((unsigned)ub.s << 16);
}

// ---------------------------------------------------------------------------
// Kernel 0: zero the num_gaussians scalar (d_out slot 2,560,000)
// ---------------------------------------------------------------------------
__global__ void gs_zero(float* cnt) {
  if (threadIdx.x == 0) *cnt = 0.0f;
}

// ---------------------------------------------------------------------------
// Kernel 1: per-gaussian conic parameters + mask count
// params[idx*8] = { u, v, -0.5*cA, -cB, -0.5*cC, op_eff, 0, 0 }
// ---------------------------------------------------------------------------
__global__ void gs_prep(const float* __restrict__ means,
                        const float* __restrict__ cov,
                        const float* __restrict__ opac,
                        float* __restrict__ prm,
                        float* __restrict__ cnt) {
  int idx = blockIdx.x * 256 + threadIdx.x;
  if (idx >= NB * NG) return;
  float op = opac[idx];
  float x = means[idx * 3 + 0];
  float y = means[idx * 3 + 1];
  // sh = sw = 200/100 = 2.0 ; centers = 100
  float u = -2.0f * y + 100.0f;
  float v = -2.0f * x + 100.0f;
  float sxx = cov[idx * 6 + 0];
  float sxy = cov[idx * 6 + 1];
  float syy = cov[idx * 6 + 3];
  float c00 = 4.0f * syy + 0.3f;
  float c01 = 4.0f * sxy;
  float c11 = 4.0f * sxx + 0.3f;
  float det = c00 * c11 - c01 * c01;
  bool mask  = op > 0.05f;
  bool valid = mask && (det > 0.0f);
  float inv_det = (det > 0.0f) ? (1.0f / det) : 0.0f;
  float cA = c11 * inv_det;
  float cB = -c01 * inv_det;
  float cC = c00 * inv_det;
  float* p = prm + idx * 8;
  p[0] = u;
  p[1] = v;
  p[2] = -0.5f * cA;
  p[3] = -cB;
  p[4] = -0.5f * cC;
  p[5] = valid ? op : 0.0f;   // op_eff == 0 kills alpha entirely
  p[6] = 0.0f;
  p[7] = 0.0f;
  if (mask) atomicAdd(cnt, 1.0f / (float)NB);  // mean over batch of counts
}

// ---------------------------------------------------------------------------
// Kernel 2: render. 8 waves/block, 32 pixels/wave (one pixel per lane).
// Per 32-gaussian chunk: per-lane alpha + transmittance scan -> f16 weights
// in LDS -> 4x v_wmma_f32_16x16x32_f16 (D=32 x 32 pixels) per wave.
// ---------------------------------------------------------------------------
__global__ void __launch_bounds__(256)
gs_render(const float* __restrict__ prm,
          const float* __restrict__ feat,
          float* __restrict__ out) {
  __shared__ float    sprm[32 * 8];     // params for 32 gaussians
  __shared__ unsigned sfeat[32 * 16];   // f16 features [d][g-pair], 2KB
  __shared__ unsigned sw[8][32 * 16];   // f16 weights  [wave][px][g-pair], 16KB

  const int tid  = threadIdx.x;
  const int wave = tid >> 5;
  const int lane = tid & 31;
  const int b    = blockIdx.y;
  const int pixBase = blockIdx.x * 256;
  const int myPix   = pixBase + wave * 32 + lane;
  const float pixOk = (myPix < NPIX) ? 1.0f : 0.0f;
  const int   pc    = (myPix < NPIX) ? myPix : 0;
  const float gi = (float)(pc / BEV_W);   // row index -> du
  const float gj = (float)(pc % BEV_W);   // col index -> dv

  float T = 1.0f;
  v8f acc[4];
  #pragma unroll
  for (int i = 0; i < 4; ++i) acc[i] = (v8f){0.f,0.f,0.f,0.f,0.f,0.f,0.f,0.f};

  const float* featB = feat + (size_t)b * NG * ND;
  const float* prmB  = prm  + (size_t)b * NG * 8;

  for (int g0 = 0; g0 < NG; g0 += 32) {
    __syncthreads();
    // stage params: 32 g * 8 floats, one per thread
    {
      int gIdx = g0 + (tid >> 3);
      sprm[tid] = (gIdx < NG) ? prmB[gIdx * 8 + (tid & 7)] : 0.0f;
    }
    // stage features as f16 pairs: sfeat[d*16 + gp] = half2(f[g],f[g+1])
    #pragma unroll
    for (int e = tid; e < 512; e += 256) {
      int d  = e >> 4;
      int gp = e & 15;
      int g  = g0 + gp * 2;
      float f0 = (g     < NG) ? featB[(size_t)g * ND + d]       : 0.0f;
      float f1 = (g + 1 < NG) ? featB[(size_t)(g + 1) * ND + d] : 0.0f;
      sfeat[e] = pack2h(f0, f1);
    }
    __syncthreads();

    // --- ordered alpha/transmittance scan for this lane's pixel ---
    float wprev = 0.0f;
    #pragma unroll 8
    for (int k = 0; k < 32; ++k) {
      const float* pf = &sprm[k * 8];
      float du = pf[0] - gi;
      float dv = pf[1] - gj;
      float pw = pf[2] * du * du + pf[4] * dv * dv + pf[3] * (du * dv);
      float al = fminf(0.99f, (pf[5] * pixOk) * __expf(pw));
      if (pw > 0.0f || al < (1.0f / 255.0f)) al = 0.0f;
      float w = al * T;
      T *= (1.0f - al);
      if (k & 1) sw[wave][lane * 16 + (k >> 1)] = pack2h(wprev, w);
      else       wprev = w;
    }
    // LDS ops are in-order within a wave: reads below see our stores.

    union UF { v16h h; unsigned u[8]; };
    UF afrag[2], bfrag[2];

    // B fragments (weights 32x16): lane -> N=lane&15, K half by lane>>4
    #pragma unroll
    for (int nt = 0; nt < 2; ++nt) {
      int pxl = nt * 16 + (lane & 15);
      int kb  = (lane >> 4) * 8;
      #pragma unroll
      for (int j = 0; j < 8; ++j)
        bfrag[nt].u[j] = sw[wave][pxl * 16 + kb + j];
    }
    // A fragments (features 16x32): lane -> M=lane&15; reg j -> K pair
    #pragma unroll
    for (int dt = 0; dt < 2; ++dt) {
      int d = dt * 16 + (lane & 15);
      #pragma unroll
      for (int j = 0; j < 8; ++j) {
        int k = ((lane >> 4) * 16) + (j * 2);
        afrag[dt].u[j] = sfeat[d * 16 + (k >> 1)];
      }
    }
    #pragma unroll
    for (int dt = 0; dt < 2; ++dt)
      #pragma unroll
      for (int nt = 0; nt < 2; ++nt)
        acc[dt * 2 + nt] = __builtin_amdgcn_wmma_f32_16x16x32_f16(
            false, afrag[dt].h, false, bfrag[nt].h,
            (short)0, acc[dt * 2 + nt], false, false);
  }

  // write out[b][d][p]; C/D layout: reg r -> M = r + 8*(lane>=16), N = lane&15
  float* outB = out + (size_t)b * ND * NPIX;
  #pragma unroll
  for (int dt = 0; dt < 2; ++dt)
    #pragma unroll
    for (int nt = 0; nt < 2; ++nt) {
      int px = pixBase + wave * 32 + nt * 16 + (lane & 15);
      if (px < NPIX) {
        #pragma unroll
        for (int r = 0; r < 8; ++r) {
          int d = dt * 16 + r + ((lane >> 4) ? 8 : 0);
          outB[(size_t)d * NPIX + px] = acc[dt * 2 + nt][r];
        }
      }
    }
}

// ---------------------------------------------------------------------------
extern "C" void kernel_launch(void* const* d_in, const int* in_sizes, int n_in,
                              void* d_out, int out_size, void* d_ws, size_t ws_size,
                              hipStream_t stream) {
  const float* features = (const float*)d_in[0];  // (B,G,32)
  const float* means3D  = (const float*)d_in[1];  // (B,G,3)
  const float* cov3D    = (const float*)d_in[2];  // (B,G,6)
  const float* opac     = (const float*)d_in[3];  // (B,G,1)
  float* out = (float*)d_out;                     // (B,32,200,200) + scalar
  float* prm = (float*)d_ws;                      // B*G*8 floats = 128 KB
  float* cnt = out + (size_t)NB * ND * NPIX;      // num_gaussians slot

  gs_zero<<<1, 1, 0, stream>>>(cnt);
  gs_prep<<<(NB * NG + 255) / 256, 256, 0, stream>>>(means3D, cov3D, opac, prm, cnt);
  dim3 grid((NPIX + 255) / 256, NB);
  gs_render<<<grid, 256, 0, stream>>>(prm, features, out);
}
